// EncoderCustomRNN_69733089018332
// MI455X (gfx1250) — matrix operand order repaired and checked by reading, not compiled, not run on hardware
//
#include <hip/hip_runtime.h>
#include <hip/hip_bf16.h>
#include <math.h>

#define VOCAB 32000
#define H 256
#define NLAYER 10
#define BATCH 256
#define TSTEPS 512
#define ROWS 32               // batch rows per workgroup
#define NRB (BATCH / ROWS)    // 8 row blocks
#define THREADS 256           // 8 waves (wave32)

typedef __attribute__((ext_vector_type(16))) __bf16 v16bf;
typedef __attribute__((ext_vector_type(8)))  float  v8f;

union V16 { uint4 q[2]; v16bf v; };

// A-operand (16x32 bf16) from row-major [rows][256] LDS.
// Lane (m = lane&15 [+16 for second M tile handled by caller], hi = lane>>4):
// elems 0..7  -> K = kbase + 8*hi + 0..7
// elems 8..15 -> K = kbase + 16 + 8*hi + 0..7
__device__ __forceinline__ v16bf load_a(const __bf16* row, int kbase, int lane) {
  const uint4* p = (const uint4*)(row + kbase + 8 * (lane >> 4));
  V16 u; u.q[0] = p[0]; u.q[1] = p[2];
  return u.v;
}

// B-operand (32x16 bf16 = K x N) from row-major W[n][k] LDS.
// Lane n = lane&15, hi = lane>>4: elems 0..7 -> K = kbase + 8*hi + 0..7, elems 8..15 -> K = kbase + 16 + 8*hi + 0..7
__device__ __forceinline__ v16bf load_b(const __bf16* row, int kbase, int lane) {
  const uint4* p = (const uint4*)(row + kbase + 8 * (lane >> 4));
  V16 u; u.q[0] = p[0]; u.q[1] = p[2];
  return u.v;
}

__device__ __forceinline__ v8f wmma_bf16(v16bf a, v16bf b, v8f c) {
  return __builtin_amdgcn_wmma_f32_16x16x32_bf16(
      /*neg_a=*/false, a, /*neg_b=*/false, b,
      /*c_mod=*/(short)0, c, /*reuse_a=*/false, /*reuse_b=*/false);
}

__device__ __forceinline__ void wait_ge(int* p, int v) {
  while (__hip_atomic_load(p, __ATOMIC_ACQUIRE, __HIP_MEMORY_SCOPE_AGENT) < v)
    __builtin_amdgcn_s_sleep(1);
}

__global__ void init_flags_kernel(int* flags, int n) {
  int i = threadIdx.x;
  if (i < n) flags[i] = 0;
}

// LDS layout (dynamic):
//   wx_lds  [256][256] bf16  : 131072 B
//   wh_lds  [256][256] bf16  : 131072 B
//   x_lds   [32][256]  bf16  :  16384 B
//   h_lds   [32][256]  bf16  :  16384 B
//   bias    [256]      f32   :   1024 B
//   total: 295936 B  (<= 320KB WGP LDS)
#define SMEM_BYTES 295936

__global__ void __launch_bounds__(THREADS, 1)
rnn_pipeline_kernel(const int* __restrict__ tokens,     // [B,T]
                    const float* __restrict__ hidden0,  // [L,B,H]
                    const float* __restrict__ emb,      // [VOCAB,H]
                    const float* __restrict__ Wx_w,     // [L,H,H]
                    const float* __restrict__ Wx_b,     // [L,H]
                    const float* __restrict__ Wh_w,     // [L,H,H]
                    const float* __restrict__ Wh_b,     // [L,H]
                    float* __restrict__ out,            // [B,T,H] then [L,B,H]
                    int* __restrict__ done,             // [L][NRB] produced count
                    int* __restrict__ ack,              // [L][NRB] consumed count
                    __bf16* __restrict__ ring)          // [L][NRB][2][32][256] bf16
{
  extern __shared__ char smem[];
  __bf16* wx_lds   = (__bf16*)(smem);
  __bf16* wh_lds   = (__bf16*)(smem + 131072);
  __bf16* x_lds    = (__bf16*)(smem + 262144);
  __bf16* h_lds    = (__bf16*)(smem + 278528);
  float*  bias_lds = (float*) (smem + 294912);

  const int rb   = blockIdx.x;   // row block 0..7
  const int l    = blockIdx.y;   // layer 0..9
  const int tid  = threadIdx.x;
  const int lane = tid & 31;
  const int wave = tid >> 5;     // 0..7

  // ---- one-time LDS fill: weights (bf16), bias, initial hidden ----
  {
    const float4* wx4 = (const float4*)(Wx_w + (size_t)l * H * H);
    const float4* wh4 = (const float4*)(Wh_w + (size_t)l * H * H);
    for (int i = tid; i < H * H / 4; i += THREADS) {
      float4 fx = wx4[i];
      float4 fh = wh4[i];
      int o = i * 4;
      wx_lds[o + 0] = (__bf16)fx.x; wx_lds[o + 1] = (__bf16)fx.y;
      wx_lds[o + 2] = (__bf16)fx.z; wx_lds[o + 3] = (__bf16)fx.w;
      wh_lds[o + 0] = (__bf16)fh.x; wh_lds[o + 1] = (__bf16)fh.y;
      wh_lds[o + 2] = (__bf16)fh.z; wh_lds[o + 3] = (__bf16)fh.w;
    }
    if (tid < H) bias_lds[tid] = Wx_b[l * H + tid] + Wh_b[l * H + tid];
    for (int i = tid; i < ROWS * H; i += THREADS) {
      int row = i >> 8, col = i & 255;
      h_lds[i] = (__bf16)hidden0[((size_t)l * BATCH + rb * ROWS + row) * H + col];
    }
  }
  __syncthreads();

  float* out_seq  = out;                          // [B,T,H]
  float* out_hfin = out + (size_t)BATCH * TSTEPS * H;  // [L,B,H]

  const int last = NLAYER - 1;

  for (int t = 0; t < TSTEPS; ++t) {
    // ================= input phase: fill x_lds (bf16 [32][256]) =================
    if (l == 0) {
      // embedding gather: thread -> (row = tid/8, 32-col segment = tid%8)
      int row = tid >> 3, seg = tid & 7;
      int b = rb * ROWS + row;
      int tok = tokens[b * TSTEPS + t];
      const float4* e = (const float4*)(emb + (size_t)tok * H + seg * 32);
      __bf16* dst = x_lds + row * H + seg * 32;
#pragma unroll
      for (int i = 0; i < 8; ++i) {
        float4 f = e[i];
        dst[i * 4 + 0] = (__bf16)f.x; dst[i * 4 + 1] = (__bf16)f.y;
        dst[i * 4 + 2] = (__bf16)f.z; dst[i * 4 + 3] = (__bf16)f.w;
      }
      __syncthreads();
    } else {
      if (tid == 0) wait_ge(done + (l - 1) * NRB + rb, t + 1);
      __syncthreads();
      const __bf16* slot =
          ring + ((((size_t)(l - 1)) * NRB + rb) * 2 + (t & 1)) * (ROWS * H);
      if (t + 1 < TSTEPS)
        __builtin_prefetch(ring + ((((size_t)(l - 1)) * NRB + rb) * 2 +
                                   ((t + 1) & 1)) * (ROWS * H), 0, 0);
      const uint4* src = (const uint4*)slot;
      uint4* dst = (uint4*)x_lds;
#pragma unroll
      for (int i = 0; i < 4; ++i) dst[tid + i * THREADS] = src[tid + i * THREADS];
      __syncthreads();
      if (tid == 0)
        __hip_atomic_store(ack + (l - 1) * NRB + rb, t + 1,
                           __ATOMIC_RELEASE, __HIP_MEMORY_SCOPE_AGENT);
    }

    // ================= compute: [32,256] = x@WxT + h@WhT (bf16 WMMA, f32 acc) ===
    v8f acc[2][2];
#pragma unroll
    for (int mt = 0; mt < 2; ++mt)
#pragma unroll
      for (int jn = 0; jn < 2; ++jn)
#pragma unroll
        for (int v = 0; v < 8; ++v) acc[mt][jn][v] = 0.0f;

    const int mlane = lane & 15;
    const __bf16* xrow0 = x_lds + mlane * H;
    const __bf16* xrow1 = x_lds + (mlane + 16) * H;
    const __bf16* hrow0 = h_lds + mlane * H;
    const __bf16* hrow1 = h_lds + (mlane + 16) * H;
    const __bf16* wxr0  = wx_lds + (wave * 32 + mlane) * H;
    const __bf16* wxr1  = wxr0 + 16 * H;
    const __bf16* whr0  = wh_lds + (wave * 32 + mlane) * H;
    const __bf16* whr1  = whr0 + 16 * H;

#pragma unroll
    for (int k = 0; k < H; k += 32) {
      v16bf a0 = load_a(xrow0, k, lane);
      v16bf a1 = load_a(xrow1, k, lane);
      v16bf b0 = load_b(wxr0, k, lane);
      v16bf b1 = load_b(wxr1, k, lane);
      acc[0][0] = wmma_bf16(a0, b0, acc[0][0]);
      acc[0][1] = wmma_bf16(a0, b1, acc[0][1]);
      acc[1][0] = wmma_bf16(a1, b0, acc[1][0]);
      acc[1][1] = wmma_bf16(a1, b1, acc[1][1]);
      v16bf c0 = load_a(hrow0, k, lane);
      v16bf c1 = load_a(hrow1, k, lane);
      v16bf d0 = load_b(whr0, k, lane);
      v16bf d1 = load_b(whr1, k, lane);
      acc[0][0] = wmma_bf16(c0, d0, acc[0][0]);
      acc[0][1] = wmma_bf16(c0, d1, acc[0][1]);
      acc[1][0] = wmma_bf16(c1, d0, acc[1][0]);
      acc[1][1] = wmma_bf16(c1, d1, acc[1][1]);
    }

    // bias + tanh (in accumulators)
#pragma unroll
    for (int mt = 0; mt < 2; ++mt)
#pragma unroll
      for (int jn = 0; jn < 2; ++jn) {
        float bsum = bias_lds[(wave * 2 + jn) * 16 + mlane];
#pragma unroll
        for (int v = 0; v < 8; ++v)
          acc[mt][jn][v] = tanhf(acc[mt][jn][v] + bsum);
      }

    // ================= output phase =================
    // producer backpressure: slot (t&1) reusable once consumer finished t-2
    if (l < last && tid == 0) wait_ge(ack + l * NRB + rb, t - 1);
    __syncthreads();  // also guarantees all compute reads of h_lds/x_lds done

    __bf16* oslot =
        ring + (((size_t)l * NRB + rb) * 2 + (t & 1)) * (ROWS * H);
    const int hi2 = lane >> 4;
#pragma unroll
    for (int mt = 0; mt < 2; ++mt)
#pragma unroll
      for (int jn = 0; jn < 2; ++jn) {
        int ncol = (wave * 2 + jn) * 16 + mlane;
#pragma unroll
        for (int v = 0; v < 8; ++v) {
          int mrow = mt * 16 + v + 8 * hi2;   // D layout: M = v + 8*(lane>=16)
          float val = acc[mt][jn][v];
          h_lds[mrow * H + ncol] = (__bf16)val;
          if (l == last)
            out_seq[((size_t)(rb * ROWS + mrow) * TSTEPS + t) * H + ncol] = val;
          else
            oslot[mrow * H + ncol] = (__bf16)val;
          if (t == TSTEPS - 1)
            out_hfin[((size_t)l * BATCH + rb * ROWS + mrow) * H + ncol] = val;
        }
      }

    if (l < last) {
      __threadfence();     // make ring writes device-visible
      __syncthreads();
      if (tid == 0)
        __hip_atomic_store(done + l * NRB + rb, t + 1,
                           __ATOMIC_RELEASE, __HIP_MEMORY_SCOPE_AGENT);
    }
  }
}

extern "C" void kernel_launch(void* const* d_in, const int* in_sizes, int n_in,
                              void* d_out, int out_size, void* d_ws, size_t ws_size,
                              hipStream_t stream) {
  const int*   tokens  = (const int*)d_in[0];
  const float* hidden0 = (const float*)d_in[1];
  const float* emb     = (const float*)d_in[2];
  const float* Wx_w    = (const float*)d_in[3];
  const float* Wx_b    = (const float*)d_in[4];
  const float* Wh_w    = (const float*)d_in[5];
  const float* Wh_b    = (const float*)d_in[6];
  float* out = (float*)d_out;

  int* done = (int*)d_ws;
  int* ack  = done + NLAYER * NRB;
  __bf16* ring = (__bf16*)(ack + NLAYER * NRB);

  // re-zero sync flags every call (harness does not re-poison between replays)
  init_flags_kernel<<<1, 256, 0, stream>>>(done, 2 * NLAYER * NRB);

  (void)hipFuncSetAttribute((const void*)rnn_pipeline_kernel,
                            hipFuncAttributeMaxDynamicSharedMemorySize,
                            SMEM_BYTES);

  dim3 grid(NRB, NLAYER);  // 80 persistent workgroups, one per (row-block, layer)
  rnn_pipeline_kernel<<<grid, THREADS, SMEM_BYTES, stream>>>(
      tokens, hidden0, emb, Wx_w, Wx_b, Wh_w, Wh_b, out, done, ack, ring);
}
